// CausalAttention_81879256531555
// MI455X (gfx1250) — matrix operand order repaired and checked by reading, not compiled, not run on hardware
//
#include <hip/hip_runtime.h>
#include <hip/hip_bf16.h>

#define D_MODEL 1024
#define NH      16
#define DH      64
#define SEQ     2048
#define NBATCH  4
#define MTOT    (NBATCH * SEQ)   // 8192 rows
#define NQKV    (3 * NH * DH)    // 3072 cols
#define L2E     1.4426950408889634f

typedef _Float16 v16h __attribute__((ext_vector_type(16)));
typedef _Float16 h8   __attribute__((ext_vector_type(8)));
typedef float    v8f  __attribute__((ext_vector_type(8)));

__device__ __forceinline__ v8f wmma_f16(v16h a, v16h b, v8f c) {
  // D = A(16x32 f16) * B(32x16 f16) + C(16x16 f32)
  return __builtin_amdgcn_wmma_f32_16x16x32_f16(false, a, false, b,
                                                (short)0, c, false, false);
}

__device__ __forceinline__ v8f vzero8() {
  v8f z;
#pragma unroll
  for (int i = 0; i < 8; ++i) z[i] = 0.0f;
  return z;
}

// Build a 16x32 A-fragment from a per-lane row pointer (K contiguous in mem).
// Layout (contiguous): lanes 0-15 (row m=lane) hold K = 0..15;
// lanes 16-31 (row m=lane-16) hold K = 16..31.
__device__ __forceinline__ v16h load_afrag(const _Float16* row, int hi) {
  v16h a;
  *(h8*)&a        = *(const h8*)(row + 16 * hi);
  *(((h8*)&a) + 1) = *(const h8*)(row + 16 * hi + 8);
  return a;
}

// ---------------------------------------------------------------------------
// Kernel 1: QKV projection.  [8192 x 1024] @ [1024 x 3072] + bias.
// Writes Q (pre-scaled by 1/8) and K as f16 [b,h,l,64]; V transposed [b,h,64,l].
// ---------------------------------------------------------------------------
__global__ __launch_bounds__(256) void qkv_gemm(
    const float* __restrict__ X, const float* __restrict__ Wqkv,
    const float* __restrict__ bqkv, _Float16* __restrict__ qws,
    _Float16* __restrict__ kws, _Float16* __restrict__ vtws) {
  __shared__ _Float16 As[128][40];  // A tile, row-major [m][k], padded
  __shared__ _Float16 Bs[128][40];  // B tile transposed  [n][k], padded

  const int t = threadIdx.x;
  const int lane = t & 31, wave = t >> 5;
  const int hi = lane >> 4, ln = lane & 15;
  const int m0 = blockIdx.y * 128;
  const int n0 = blockIdx.x * 128;

  v8f acc[8];
#pragma unroll
  for (int i = 0; i < 8; ++i) acc[i] = vzero8();

  for (int k0 = 0; k0 < D_MODEL; k0 += 32) {
    // Stage A: each thread converts 16 f32 -> f16 (coalesced float4 loads)
    {
      const int r = t >> 1, c = (t & 1) * 16;
      const float4* src =
          (const float4*)(X + (size_t)(m0 + r) * D_MODEL + k0 + c);
      _Float16* dst = &As[r][c];
#pragma unroll
      for (int i = 0; i < 4; ++i) {
        float4 v = src[i];
        dst[i * 4 + 0] = (_Float16)v.x;
        dst[i * 4 + 1] = (_Float16)v.y;
        dst[i * 4 + 2] = (_Float16)v.z;
        dst[i * 4 + 3] = (_Float16)v.w;
      }
    }
    // Stage B transposed: coalesced along n in global, scatter [n][k] in LDS
    {
      const int nn = t & 127;
      const int kb = t >> 7;  // 0 or 1
#pragma unroll
      for (int i = 0; i < 16; ++i) {
        const int kk = kb + 2 * i;
        Bs[nn][kk] = (_Float16)Wqkv[(size_t)(k0 + kk) * NQKV + n0 + nn];
      }
    }
    __syncthreads();

    // B frag for this wave's n-subtile: lane n holds K = e + 16*hi contiguous
    const v16h bfrag = *(const v16h*)&Bs[wave * 16 + ln][16 * hi];
#pragma unroll
    for (int ms = 0; ms < 8; ++ms) {
      const v16h afrag = load_afrag(&As[ms * 16 + ln][0], hi);
      acc[ms] = wmma_f16(afrag, bfrag, acc[ms]);
    }
    __syncthreads();
  }

  // Epilogue: bias, split into Q/K/V, scale Q by 1/sqrt(64)
  const int n_g = n0 + wave * 16 + ln;
  const float bias = bqkv[n_g];
  const int h = n_g / 192;
  const int rem = n_g % 192;
  const int which = rem / 64;  // 0=Q, 1=K, 2=V
  const int d = rem % 64;
#pragma unroll
  for (int ms = 0; ms < 8; ++ms) {
#pragma unroll
    for (int r = 0; r < 8; ++r) {
      const int m_g = m0 + ms * 16 + r + 8 * hi;
      const int b = m_g >> 11;       // / SEQ
      const int l = m_g & (SEQ - 1);
      const float v = acc[ms][r] + bias;
      const size_t bh = (size_t)b * NH + h;
      if (which == 0) {
        qws[(bh * SEQ + l) * DH + d] = (_Float16)(v * 0.125f);
      } else if (which == 1) {
        kws[(bh * SEQ + l) * DH + d] = (_Float16)v;
      } else {
        vtws[(bh * DH + d) * SEQ + l] = (_Float16)v;
      }
    }
  }
}

// ---------------------------------------------------------------------------
// Kernel 2: causal flash attention.  One WG per (b*h, 128-query block);
// each wave owns 16 queries (no cross-wave sync -> causal divergence is fine).
// ---------------------------------------------------------------------------
__global__ __launch_bounds__(256) void attn_kernel(
    const _Float16* __restrict__ qws, const _Float16* __restrict__ kws,
    const _Float16* __restrict__ vtws, _Float16* __restrict__ attnws) {
  __shared__ _Float16 Ps[8][16][40];  // per-wave P scratch (16 q x 32 k)

  const int t = threadIdx.x;
  const int lane = t & 31, wave = t >> 5;
  const int hi = lane >> 4, ln = lane & 15;
  const int bh = blockIdx.y;
  const int q0 = blockIdx.x * 128 + wave * 16;
  const size_t base = (size_t)bh * SEQ * DH;

  // Q fragments (already scaled by 1/8): rows q0..q0+15, d split 0..31/32..63
  const _Float16* qrow = qws + base + (size_t)(q0 + ln) * DH;
  const v16h qa0 = load_afrag(qrow, hi);
  const v16h qa1 = load_afrag(qrow + 32, hi);

  v8f o[4];
#pragma unroll
  for (int j = 0; j < 4; ++j) o[j] = vzero8();
  float Mrow[8], Lrow[8];
#pragma unroll
  for (int r = 0; r < 8; ++r) { Mrow[r] = -3.0e38f; Lrow[r] = 0.0f; }

  const int nkeys = q0 + 16;
  for (int kc = 0; kc < nkeys; kc += 32) {
    // S = Q * K^T for 32 keys (two 16-key subtiles)
    v8f s0 = vzero8(), s1 = vzero8();
    const _Float16* krow0 = kws + base + (size_t)(kc + ln) * DH;
    const _Float16* krow1 = krow0 + 16 * DH;
    s0 = wmma_f16(qa0, *(const v16h*)(krow0 + 16 * hi), s0);
    s0 = wmma_f16(qa1, *(const v16h*)(krow0 + 32 + 16 * hi), s0);
    s1 = wmma_f16(qa0, *(const v16h*)(krow1 + 16 * hi), s1);
    s1 = wmma_f16(qa1, *(const v16h*)(krow1 + 32 + 16 * hi), s1);

    // Causal mask + online softmax in C-fragment layout (row = r + 8*hi)
    float rmax[8], rsum[8];
#pragma unroll
    for (int r = 0; r < 8; ++r) {
      const int q = q0 + r + 8 * hi;
      if (kc + ln > q) s0[r] = -3.0e38f;
      if (kc + 16 + ln > q) s1[r] = -3.0e38f;
      rmax[r] = fmaxf(s0[r], s1[r]);
    }
#pragma unroll
    for (int off = 1; off <= 8; off <<= 1) {
#pragma unroll
      for (int r = 0; r < 8; ++r)
        rmax[r] = fmaxf(rmax[r], __shfl_xor(rmax[r], off, 32));
    }
#pragma unroll
    for (int r = 0; r < 8; ++r) {
      const float mnew = fmaxf(Mrow[r], rmax[r]);
      const float corr = exp2f((Mrow[r] - mnew) * L2E);
      const float p0 = exp2f((s0[r] - mnew) * L2E);
      const float p1 = exp2f((s1[r] - mnew) * L2E);
      s0[r] = p0;
      s1[r] = p1;
      rsum[r] = p0 + p1;
      Mrow[r] = mnew;
      Lrow[r] = Lrow[r] * corr;
#pragma unroll
      for (int j = 0; j < 4; ++j) o[j][r] *= corr;
    }
#pragma unroll
    for (int off = 1; off <= 8; off <<= 1) {
#pragma unroll
      for (int r = 0; r < 8; ++r) rsum[r] += __shfl_xor(rsum[r], off, 32);
    }
#pragma unroll
    for (int r = 0; r < 8; ++r) Lrow[r] += rsum[r];

    // C-layout -> A-layout for P via per-wave LDS (same-wave DS is in-order)
#pragma unroll
    for (int r = 0; r < 8; ++r) {
      Ps[wave][r + 8 * hi][ln] = (_Float16)s0[r];
      Ps[wave][r + 8 * hi][16 + ln] = (_Float16)s1[r];
    }
    __threadfence_block();
    const v16h pfrag = load_afrag(&Ps[wave][ln][0], hi);

    // O += P * V  (V^T layout: lane's column d is contiguous over keys)
    const _Float16* vb = vtws + base;
#pragma unroll
    for (int j = 0; j < 4; ++j) {
      const _Float16* vrow =
          vb + (size_t)(j * 16 + ln) * SEQ + kc + 16 * hi;
      o[j] = wmma_f16(pfrag, *(const v16h*)vrow, o[j]);
    }
    __threadfence_block();  // keep next-iter Ps stores after this iter's loads
  }

  // Normalize, write f16 attention output [b, l, h*64]
  const int h = bh & (NH - 1);
  const int b = bh >> 4;
#pragma unroll
  for (int j = 0; j < 4; ++j) {
#pragma unroll
    for (int r = 0; r < 8; ++r) {
      const int l = q0 + r + 8 * hi;
      const float v = o[j][r] / Lrow[r];
      attnws[((size_t)(b * SEQ + l)) * (NH * DH) + h * DH + j * 16 + ln] =
          (_Float16)v;
    }
  }
}

// ---------------------------------------------------------------------------
// Kernel 3: output projection.  [8192 x 1024](f16) @ [1024 x 1024] + bias.
// A-tile staging uses GLOBAL_LOAD_ASYNC_TO_LDS_B128 (CDNA5 async copy,
// tracked on ASYNCcnt): pure f16 copy, no VGPR round-trip.
// ---------------------------------------------------------------------------
__global__ __launch_bounds__(256) void out_gemm(
    const _Float16* __restrict__ Aws, const float* __restrict__ Wout,
    const float* __restrict__ bout, float* __restrict__ out) {
  __shared__ _Float16 As[128][40];
  __shared__ _Float16 Bs[128][40];

  const int t = threadIdx.x;
  const int lane = t & 31, wave = t >> 5;
  const int hi = lane >> 4, ln = lane & 15;
  const int m0 = blockIdx.y * 128;
  const int n0 = blockIdx.x * 128;

  // Per-thread async copy addresses: thread t owns row t/2, 16-half chunk.
  const int ar = t >> 1, ac = (t & 1) * 16;
  const unsigned int lds_dst = (unsigned int)(size_t)&As[ar][ac];
  unsigned long long gsrc =
      (unsigned long long)(size_t)(Aws + (size_t)(m0 + ar) * D_MODEL + ac);

  v8f acc[8];
#pragma unroll
  for (int i = 0; i < 8; ++i) acc[i] = vzero8();

  for (int k0 = 0; k0 < D_MODEL; k0 += 32) {
    // Stage A tile via async global->LDS copy (2 x 16B per thread).
    // INST_OFFSET is added to BOTH the LDS and global addresses (ISA 15.18).
    asm volatile(
        "global_load_async_to_lds_b128 %0, %1, off offset:0\n\t"
        "global_load_async_to_lds_b128 %0, %1, off offset:16"
        :
        : "v"(lds_dst), "v"(gsrc)
        : "memory");
    gsrc += 32 * sizeof(_Float16);  // advance along k for next iteration

    // Stage B transposed (f32 -> f16 conversion, cannot use async path)
    {
      const int nn = t & 127;
      const int kb = t >> 7;
#pragma unroll
      for (int i = 0; i < 16; ++i) {
        const int kk = kb + 2 * i;
        Bs[nn][kk] = (_Float16)Wout[(size_t)(k0 + kk) * D_MODEL + n0 + nn];
      }
    }
    asm volatile("s_wait_asynccnt 0x0" ::: "memory");
    __syncthreads();

    const v16h bfrag = *(const v16h*)&Bs[wave * 16 + ln][16 * hi];
#pragma unroll
    for (int ms = 0; ms < 8; ++ms) {
      const v16h afrag = load_afrag(&As[ms * 16 + ln][0], hi);
      acc[ms] = wmma_f16(afrag, bfrag, acc[ms]);
    }
    __syncthreads();
  }

  const int n_g = n0 + wave * 16 + ln;
  const float bias = bout[n_g];
#pragma unroll
  for (int ms = 0; ms < 8; ++ms) {
#pragma unroll
    for (int r = 0; r < 8; ++r) {
      const int m_g = m0 + ms * 16 + r + 8 * hi;
      out[(size_t)m_g * D_MODEL + n_g] = acc[ms][r] + bias;
    }
  }
}

// ---------------------------------------------------------------------------
extern "C" void kernel_launch(void* const* d_in, const int* in_sizes, int n_in,
                              void* d_out, int out_size, void* d_ws,
                              size_t ws_size, hipStream_t stream) {
  const float* X = (const float*)d_in[0];
  const float* Wqkv = (const float*)d_in[1];
  const float* bqkv = (const float*)d_in[2];
  const float* Wout = (const float*)d_in[3];
  const float* bout = (const float*)d_in[4];
  float* out = (float*)d_out;

  const size_t per = (size_t)NBATCH * NH * SEQ * DH;  // 8,388,608 halves
  _Float16* qws = (_Float16*)d_ws;
  _Float16* kws = qws + per;
  _Float16* vtws = kws + per;
  _Float16* attnws = vtws + per;

  qkv_gemm<<<dim3(NQKV / 128, MTOT / 128), 256, 0, stream>>>(X, Wqkv, bqkv,
                                                             qws, kws, vtws);
  attn_kernel<<<dim3(SEQ / 128, NBATCH * NH), 256, 0, stream>>>(qws, kws, vtws,
                                                                attnws);
  out_gemm<<<dim3(D_MODEL / 128, MTOT / 128), 256, 0, stream>>>(attnws, Wout,
                                                                bout, out);
}
